// DownModel_11888469475771
// MI455X (gfx1250) — compile-verified, matrix operand order checked by reading, not run on hardware
//
#include <hip/hip_runtime.h>
#include <hip/hip_bf16.h>
#include <math.h>

// ---------------- model dims (fixed by reference) ----------------
#define NROW 8192
#define HDIM 256
#define EDGE 262144
#define KDIM 512          // 2*H, gram K
#define NCLS 40
#define TOPK 16
// ALPHA = 0.5 -> both adj and re_adj weighted 0.5

typedef __attribute__((ext_vector_type(16))) __bf16 v16bf;
typedef __attribute__((ext_vector_type(8)))  float  v8f;

struct alignas(16) U128 { unsigned int a, b, c, d; };
union Frag { U128 q[2]; v16bf v; };   // trivially constructible union -> ok in device code

__device__ inline void atomAddF(float* p, float v) {
  // agent-scope relaxed fadd -> lowers to global_atomic_add_f32 on CDNA5
  __hip_atomic_fetch_add(p, v, __ATOMIC_RELAXED, __HIP_MEMORY_SCOPE_AGENT);
}

__device__ inline unsigned short f2bf(float x) {  // RNE float -> bf16
  unsigned int u = __float_as_uint(x);
  unsigned int r = u + 0x7FFFu + ((u >> 16) & 1u);
  return (unsigned short)(r >> 16);
}

// ---------------- tiny / elementwise kernels ----------------
__global__ void k_zero(float* __restrict__ p, int n) {
  int i = blockIdx.x * blockDim.x + threadIdx.x;
  if (i < n) p[i] = 0.f;
}

__global__ void k_pre_token(const float* __restrict__ wp, const float* __restrict__ tokens,
                            float* __restrict__ pre) {
  int h = threadIdx.x;  // 256 threads
  float s = 0.f;
  #pragma unroll
  for (int t = 0; t < 4; ++t) s += wp[t] * tokens[t * HDIM + h];
  pre[h] = s;
}

__global__ __launch_bounds__(256) void k_features1(
    const float* __restrict__ feat, const float* __restrict__ pre,
    const float* __restrict__ gtok, const float* __restrict__ ptw,
    const float* __restrict__ cw, float* __restrict__ f1) {
  int idx = blockIdx.x * blockDim.x + threadIdx.x;          // N*H threads
  int h = idx & (HDIM - 1);
  float f  = feat[idx];
  float x  = pre[h] * f; x = x > 0.f ? x : 0.f; x *= gtok[h];
  float x1 = ptw[h] * f;
  float z  = cw[0] * x + cw[1] * x1;
  f1[idx] = z > 0.f ? z : (expf(z) - 1.f);                  // ELU(alpha=1)
}

// agg[rows[e]] += vals[e] * f1[cols[e]]  -- 4 floats per thread
__global__ __launch_bounds__(256) void k_spmm_agg(
    const int* __restrict__ rows, const int* __restrict__ cols,
    const float* __restrict__ vals, const float* __restrict__ f1,
    float* __restrict__ agg) {
  int idx = blockIdx.x * blockDim.x + threadIdx.x;          // E*64 threads
  int e = idx >> 6, q = (idx & 63) << 2;
  float a = vals[e];
  const float4 f = *reinterpret_cast<const float4*>(&f1[(size_t)cols[e] * HDIM + q]);
  float* p = &agg[(size_t)rows[e] * HDIM + q];
  atomAddF(p + 0, a * f.x);
  atomAddF(p + 1, a * f.y);
  atomAddF(p + 2, a * f.z);
  atomAddF(p + 3, a * f.w);
}

// row-normalize balance_w * [f1 | agg] -> bf16 emb (L2-resident, 8 MB)
__global__ __launch_bounds__(256) void k_emb(
    const float* __restrict__ f1, const float* __restrict__ agg,
    const float* __restrict__ bal, unsigned short* __restrict__ emb16) {
  __shared__ float red[256];
  int n = blockIdx.x, tid = threadIdx.x;
  float v0 = bal[tid]        * f1 [(size_t)n * HDIM + tid];
  float v1 = bal[tid + HDIM] * agg[(size_t)n * HDIM + tid];
  red[tid] = v0 * v0 + v1 * v1;
  __syncthreads();
  for (int s = 128; s > 0; s >>= 1) {
    if (tid < s) red[tid] += red[tid + s];
    __syncthreads();
  }
  float inv = 1.f / (sqrtf(red[0]) + 1e-8f);
  emb16[(size_t)n * KDIM + tid]        = f2bf(v0 * inv);
  emb16[(size_t)n * KDIM + tid + HDIM] = f2bf(v1 * inv);
}

// ---------------- fused Gram (emb @ emb^T) + top-16 ----------------
// grid = 256 blocks (one per 32-row M block), block = 256 (8 wave32).
// Each wave register-blocks 2 M-tiles x 2 N-tiles: every A frag (LDS) and
// every B frag (global/L2) feeds 2 WMMAs -> 1 ds_b128 + 1 global_b128 per wmma,
// halving L2 traffic (2 GB total, ~34 FLOP per L2 byte).
__global__ __launch_bounds__(256) void k_gram_topk(
    const unsigned short* __restrict__ emb16,
    float* __restrict__ tv, int* __restrict__ ti) {
  __shared__ alignas(16) unsigned short As[32 * KDIM];  // 32 KB: 32-row A block
  __shared__ float spill[8][32 * 32];                   // 32 KB: per-wave 32x32 tile
  __shared__ float lv[8][32][TOPK];                     // 16 KB: per-wave top-16
  __shared__ int   li[8][32][TOPK];                     // 16 KB

  const int tid  = threadIdx.x;
  const int wv   = tid >> 5;
  const int lane = tid & 31;
  const int mbase = blockIdx.x * 32;

  { // cooperative A-block copy: 32 rows x 512 bf16 = 2048 x 16B
    const U128* src = reinterpret_cast<const U128*>(emb16 + (size_t)mbase * KDIM);
    U128* dst = reinterpret_cast<U128*>(As);
    for (int i = tid; i < 2048; i += 256) dst[i] = src[i];
  }
  { // init top-k lists
    float* lvf = &lv[0][0][0];
    int*   lif = &li[0][0][0];
    for (int i = tid; i < 8 * 32 * TOPK; i += 256) { lvf[i] = -1e30f; lif[i] = -1; }
  }
  __syncthreads();

  const int r    = lane & 15;
  const int half = lane >> 4;
  float minval = -1e30f; int minpos = 0;

  for (int t = 0; t < 32; ++t) {                  // uniform trip count, all 8 waves
    const int p = wv + (t << 3);                  // N-pair index: columns [p*32, p*32+32)
    const int cbase = p << 5;
    const unsigned short* B0 = emb16 + (size_t)(cbase + r) * KDIM;       // N-tile 0
    const unsigned short* B1 = B0 + (size_t)16 * KDIM;                   // N-tile 1
    __builtin_prefetch(B0, 0, 3);
    __builtin_prefetch(B1, 0, 3);
    v8f c00 = {0.f,0.f,0.f,0.f,0.f,0.f,0.f,0.f};
    v8f c01 = c00, c10 = c00, c11 = c00;
    for (int kk = 0; kk < 16; ++kk) {             // K = 512 = 16 x 32
      const int koff = (kk << 5) + (half << 3);   // ISA 7.12.2 16-bit A/B lane layout
      Frag a0, a1, b0, b1;
      a0.q[0] = *reinterpret_cast<const U128*>(&As[r * KDIM + koff]);
      a0.q[1] = *reinterpret_cast<const U128*>(&As[r * KDIM + koff + 16]);
      a1.q[0] = *reinterpret_cast<const U128*>(&As[(16 + r) * KDIM + koff]);
      a1.q[1] = *reinterpret_cast<const U128*>(&As[(16 + r) * KDIM + koff + 16]);
      b0.q[0] = *reinterpret_cast<const U128*>(B0 + koff);
      b0.q[1] = *reinterpret_cast<const U128*>(B0 + koff + 16);
      b1.q[0] = *reinterpret_cast<const U128*>(B1 + koff);
      b1.q[1] = *reinterpret_cast<const U128*>(B1 + koff + 16);
      c00 = __builtin_amdgcn_wmma_f32_16x16x32_bf16(false, a0.v, false, b0.v, (short)0, c00, false, false);
      c01 = __builtin_amdgcn_wmma_f32_16x16x32_bf16(false, a0.v, false, b1.v, (short)0, c01, false, false);
      c10 = __builtin_amdgcn_wmma_f32_16x16x32_bf16(false, a1.v, false, b0.v, (short)0, c10, false, false);
      c11 = __builtin_amdgcn_wmma_f32_16x16x32_bf16(false, a1.v, false, b1.v, (short)0, c11, false, false);
    }
    // spill 2x2 tile grid: acc(mi,ni) VGPR j -> row mi*16 + j + half*8, col ni*16 + r
    #pragma unroll
    for (int j = 0; j < 8; ++j) {
      const int rw0 = (j + half * 8) * 32;
      spill[wv][rw0 + r]            = c00[j];
      spill[wv][rw0 + 16 + r]       = c01[j];
      spill[wv][rw0 + 512 + r]      = c10[j];   // +16 rows
      spill[wv][rw0 + 512 + 16 + r] = c11[j];
    }
    __syncthreads();
    { // lane owns row `lane` of the 32-row block: scan 32 new candidate columns
      for (int cc = 0; cc < 32; ++cc) {
        float v = spill[wv][lane * 32 + cc];
        if (v > minval) {
          lv[wv][lane][minpos] = v;
          li[wv][lane][minpos] = cbase + cc;
          minval = lv[wv][lane][0]; minpos = 0;
          #pragma unroll
          for (int s = 1; s < TOPK; ++s) {
            float q = lv[wv][lane][s];
            if (q < minval) { minval = q; minpos = s; }
          }
        }
      }
    }
    __syncthreads();
  }

  // merge 8 waves x 16 candidates -> global top-16 per row, relu on values
  if (tid < 32) {
    const int row = mbase + tid;
    for (int s = 0; s < TOPK; ++s) {
      float best = -1e30f; int bw = 0, bs = 0;
      for (int w = 0; w < 8; ++w)
        for (int u = 0; u < TOPK; ++u) {
          float q = lv[w][tid][u];
          if (q > best) { best = q; bw = w; bs = u; }
        }
      tv[row * TOPK + s] = best > 0.f ? best : 0.f;
      ti[row * TOPK + s] = li[bw][tid][bs];
      lv[bw][tid][bs] = -1e31f;
    }
  }
}

// ---------------- sparse GCN tail ----------------
__global__ __launch_bounds__(256) void k_Y(
    const float* __restrict__ f1, const float* __restrict__ gw, float* __restrict__ Y) {
  int idx = blockIdx.x * blockDim.x + threadIdx.x;   // N*40
  int n = idx / NCLS, c = idx % NCLS;
  const float* fr = f1 + (size_t)n * HDIM;
  float s = 0.f;
  for (int h = 0; h < HDIM; ++h) s += fr[h] * gw[h * NCLS + c];
  Y[idx] = s;
}

__global__ __launch_bounds__(256) void k_edge_Z(
    const int* __restrict__ rows, const int* __restrict__ cols,
    const float* __restrict__ vals, const float* __restrict__ Y,
    float* __restrict__ Z) {
  int idx = blockIdx.x * blockDim.x + threadIdx.x;   // E*40
  int e = idx / NCLS, c = idx % NCLS;
  atomAddF(&Z[(size_t)rows[e] * NCLS + c], 0.5f * vals[e] * Y[(size_t)cols[e] * NCLS + c]);
}

__global__ __launch_bounds__(256) void k_final(
    const float* __restrict__ Z, const float* __restrict__ Y,
    const float* __restrict__ tv, const int* __restrict__ ti,
    const float* __restrict__ bias, float* __restrict__ out) {
  int idx = blockIdx.x * blockDim.x + threadIdx.x;   // N*40
  int n = idx / NCLS, c = idx % NCLS;
  float s = Z[idx] + bias[c];
  #pragma unroll
  for (int j = 0; j < TOPK; ++j)
    s += 0.5f * tv[n * TOPK + j] * Y[(size_t)ti[n * TOPK + j] * NCLS + c];
  out[idx] = s > 0.f ? s : 0.f;
}

// ---------------- launcher ----------------
extern "C" void kernel_launch(void* const* d_in, const int* in_sizes, int n_in,
                              void* d_out, int out_size, void* d_ws, size_t ws_size,
                              hipStream_t stream) {
  const float* feat  = (const float*)d_in[0];
  const int*   rows  = (const int*)  d_in[1];
  const int*   cols  = (const int*)  d_in[2];
  const float* vals  = (const float*)d_in[3];
  /* d_in[4] = down_k (hardcoded 16) */
  const float* toks  = (const float*)d_in[5];
  const float* wp    = (const float*)d_in[6];
  const float* gtok  = (const float*)d_in[7];
  const float* ptw   = (const float*)d_in[8];
  const float* cw    = (const float*)d_in[9];
  const float* bal   = (const float*)d_in[10];
  const float* gw    = (const float*)d_in[11];
  const float* gb    = (const float*)d_in[12];
  float* out = (float*)d_out;

  // workspace layout (floats), ~27.5 MB total
  float* ws   = (float*)d_ws;
  float* f1   = ws;                                  // N*H   = 2,097,152
  float* agg  = ws + 2097152;                        // N*H   = 2,097,152
  float* Z    = ws + 4194304;                        // N*40  =   327,680
  float* Y    = ws + 4521984;                        // N*40  =   327,680
  float* pre  = ws + 4849664;                        //           256
  float* tv   = ws + 4849920;                        // N*16  =   131,072
  int*   ti   = (int*)(ws + 4980992);                // N*16  =   131,072
  unsigned short* emb16 = (unsigned short*)(ws + 5112064); // N*512 bf16 (16B aligned)

  // agg and Z are contiguous -> one zeroing pass (2,424,832 floats)
  k_zero<<<(2424832 + 255) / 256, 256, 0, stream>>>(agg, 2424832);
  k_pre_token<<<1, 256, 0, stream>>>(wp, toks, pre);
  k_features1<<<(NROW * HDIM) / 256, 256, 0, stream>>>(feat, pre, gtok, ptw, cw, f1);
  k_spmm_agg<<<(EDGE * 64) / 256, 256, 0, stream>>>(rows, cols, vals, f1, agg);
  k_emb<<<NROW, 256, 0, stream>>>(f1, agg, bal, emb16);
  k_gram_topk<<<NROW / 32, 256, 0, stream>>>(emb16, tv, ti);
  k_Y<<<(NROW * NCLS) / 256, 256, 0, stream>>>(f1, gw, Y);
  k_edge_Z<<<(EDGE * NCLS) / 256, 256, 0, stream>>>(rows, cols, vals, Y, Z);
  k_final<<<(NROW * NCLS) / 256, 256, 0, stream>>>(Z, Y, tv, ti, gb, out);
}
